// SpeakerAwareCTC_32530082300567
// MI455X (gfx1250) — matrix operand order repaired and checked by reading, not compiled, not run on hardware
//
#include <hip/hip_runtime.h>
#include <hip/hip_bf16.h>
#include <math.h>

#define Bn 16
#define Tn 1000
#define Dn 512
#define Vn 5000
#define Un 64
#define Sn 129          // 2*U+1
#define VPAD 5008       // V rounded up to 16
#define NTILES 313      // ceil(V/16)
#define FINF __builtin_inff()

typedef __attribute__((ext_vector_type(16))) _Float16 v16h;
typedef __attribute__((ext_vector_type(8)))  _Float16 v8h;
typedef __attribute__((ext_vector_type(8)))  float    v8f;

// ---------- helpers ----------

// streaming logsumexp merge with -inf guard (cold paths / reductions)
__device__ __forceinline__ void lse_merge(float& M, float& S, float m2, float s2) {
    if (m2 == -FINF) return;
    if (m2 > M) { S = S * expf(M - m2) + s2; M = m2; }  // expf(-inf)=0 handles M=-inf
    else        { S += s2 * expf(m2 - M); }
}

// safe logaddexp: both -inf -> -inf, else standard
__device__ __forceinline__ float safe_la(float a, float b) {
    float mx = fmaxf(a, b);
    if (mx == -FINF) return -FINF;
    float d = fminf(a, b) - mx;          // may be -inf -> expf = 0
    return mx + log1pf(expf(d));
}

__device__ __forceinline__ bool is_inf_f(float x) { return fabsf(x) == FINF; }

// faithful port of the reference _log_sub_exp
__device__ __forceinline__ float log_sub_exp(float a, float b) {
    bool inf_a = is_inf_f(a), inf_b = is_inf_f(b);
    bool mask1 = (!inf_a) && (!inf_b);
    float a_ = mask1 ? a : -1.0f;
    float b_ = mask1 ? b : -2.0f;
    float ans_tmp = b_ + logf(expf(a_ - b_) - 1.0f);
    if (is_inf_f(ans_tmp)) { a_ = -2000.0f; b_ = -2001.0f; }
    float ans1 = b_ + logf(expf(a_ - b_) - 1.0f);
    float ans = mask1 ? ans1 : -FINF;
    bool mask2 = (!inf_a) && inf_b;
    return mask2 ? a : ans;
}

// ---------- kernel 0: W [D,V] f32 -> Wt [VPAD][D] f16 (transpose+convert, zero pad)
//            + bias [V] f32 -> biasp [VPAD] f32 (zero pad) ----------
__global__ __launch_bounds__(256) void prep_wt_kernel(const float* __restrict__ W,
                                                      const float* __restrict__ bias,
                                                      _Float16* __restrict__ Wt,
                                                      float* __restrict__ biasp) {
    int idx = blockIdx.x * 256 + threadIdx.x;   // idx = d*VPAD + v  (coalesced reads)
    if (idx >= VPAD * Dn) return;
    int d = idx / VPAD;
    int v = idx - d * VPAD;
    float w = (v < Vn) ? W[(size_t)d * Vn + v] : 0.0f;
    Wt[(size_t)v * Dn + d] = (_Float16)w;
    if (idx < VPAD) biasp[idx] = (idx < Vn) ? bias[idx] : 0.0f;
}

// ---------- kernel 1: WMMA GEMM [16 rows x 512] x [512 x V] with fused streaming logsumexp ----------
__global__ __launch_bounds__(256) void gemm_lse_kernel(const float* __restrict__ hs,
                                                       const _Float16* __restrict__ Wt,
                                                       const float* __restrict__ biasp,
                                                       float* __restrict__ lse) {
    __shared__ __align__(32) _Float16 As[16 * Dn];   // 16 KB
    __shared__ float redM[8][32][8];                 // 8 KB
    __shared__ float redS[8][32][8];                 // 8 KB

    const int tid = threadIdx.x;
    const int grow0 = blockIdx.x * 16;               // 1000 blocks x 16 rows = B*T rows

    // stage A (16 contiguous rows of hs), convert f32 -> f16
    const float4* src = (const float4*)(hs + (size_t)grow0 * Dn);
    for (int j = tid; j < 16 * Dn / 4; j += 256) {
        float4 f = src[j];
        int o = j * 4;
        As[o + 0] = (_Float16)f.x; As[o + 1] = (_Float16)f.y;
        As[o + 2] = (_Float16)f.z; As[o + 3] = (_Float16)f.w;
    }
    __syncthreads();

    const int wave = tid >> 5, lane = tid & 31;
    const int half = lane >> 4, m = lane & 15;       // wave32: two half-groups of 16 lanes
    const _Float16* arow = As + m * Dn;

    float runM[8], runS[8];
#pragma unroll
    for (int r = 0; r < 8; ++r) { runM[r] = -FINF; runS[r] = 0.0f; }

    for (int tile = wave; tile < NTILES; tile += 8) {
        const int c = tile * 16 + m;                         // this lane's column
        const _Float16* brow = Wt + (size_t)c * Dn;
        if (tile + 8 < NTILES)                               // global_prefetch_b8 next W tile
            __builtin_prefetch(Wt + (size_t)(c + 128) * Dn, 0, 1);

        v8f acc = {0.f, 0.f, 0.f, 0.f, 0.f, 0.f, 0.f, 0.f};
#pragma unroll 4
        for (int kk = 0; kk < Dn / 32; ++kk) {
            // A fragment (16x32 f16): lanes 0-15 -> K {0..7,16..23}, lanes 16-31 -> K {8..15,24..31}
            union { v16h v; v8h h[2]; } au;
            au.h[0] = *(const v8h*)(arow + kk * 32 + half * 8);
            au.h[1] = *(const v8h*)(arow + kk * 32 + 16 + half * 8);
            // B fragment (32x16 f16): lanes 0-15 -> K {0..7,16..23}, lanes 16-31 -> K {8..15,24..31}
            union { v16h v; v8h h[2]; } bu;
            bu.h[0] = *(const v8h*)(brow + kk * 32 + half * 8);
            bu.h[1] = *(const v8h*)(brow + kk * 32 + 16 + half * 8);
            v16h bfrag = bu.v;
            acc = __builtin_amdgcn_wmma_f32_16x16x32_f16(
                false, au.v, false, bfrag, (short)0, acc, false, false);
        }

        // branchless streaming-logsumexp epilogue (v_cndmask + v_max + 2x v_exp per row;
        // TRANS ops co-execute with the matrix pipe, no exec-mask branches)
        const float bc = biasp[c];
        const bool valid = (c < Vn);
#pragma unroll
        for (int r = 0; r < 8; ++r) {
            float lg = acc[r] + bc;                  // logits for row (half*8 + r), column c
            lg = valid ? lg : -FINF;                 // padded columns contribute exp(-inf)=0
            float nm = fmaxf(runM[r], lg);
            runS[r] = runS[r] * __expf(runM[r] - nm) + __expf(lg - nm);
            runM[r] = nm;
            // note: an all-invalid lane yields runM=-inf and runS=NaN; the final merge
            // skips entries with m==-inf, so the NaN never propagates.
        }
    }

#pragma unroll
    for (int r = 0; r < 8; ++r) { redM[wave][lane][r] = runM[r]; redS[wave][lane][r] = runS[r]; }
    __syncthreads();

    if (tid < 16) {                                  // thread i finalizes output row i
        const int slot = tid & 7;
        const int lbase = (tid < 8) ? 0 : 16;        // C/D layout: lanes 0-15 -> M=r, 16-31 -> M=8+r
        float M = -FINF, S = 0.0f;
        for (int w = 0; w < 8; ++w)
            for (int l = 0; l < 16; ++l)
                lse_merge(M, S, redM[w][lbase + l][slot], redS[w][lbase + l][slot]);
        lse[grow0 + tid] = M + logf(S);
    }
}

// ---------- kernel 2: gathered logits glogit[b,t,u], u=0 blank, u=1..64 labels ----------
__global__ __launch_bounds__(128) void gather_logits_kernel(const float* __restrict__ hs,
                                                            const _Float16* __restrict__ Wt,
                                                            const float* __restrict__ bias,
                                                            const int* __restrict__ ys,
                                                            float* __restrict__ glogit) {
    __shared__ float hrow[Dn];
    const int bt = blockIdx.x;
    const int b = bt / Tn;
    const float* h = hs + (size_t)bt * Dn;
    for (int j = threadIdx.x; j < Dn; j += 128) hrow[j] = h[j];
    __syncthreads();

    const int u = threadIdx.x;
    if (u < Un + 1) {
        const int c = (u == 0) ? 0 : ys[b * Un + (u - 1)];
        const _Float16* w = Wt + (size_t)c * Dn;
        float acc = 0.0f;
#pragma unroll 8
        for (int d = 0; d < Dn; ++d) acc += hrow[d] * (float)w[d];
        glogit[(size_t)bt * 65 + u] = acc + bias[c];
    }
}

// lp(b,t,s) = glogit[.., s odd ? 1+(s>>1) : 0] - lse
__device__ __forceinline__ float lp_of(const float* gl, int s) {
    return gl[(s & 1) ? (1 + (s >> 1)) : 0] - gl[65];
}

// ---------- kernel 3: alpha forward scan (one block per batch) ----------
__global__ __launch_bounds__(256) void alpha_kernel(const float* __restrict__ glogit,
                                                    const float* __restrict__ lse,
                                                    const int* __restrict__ ys,
                                                    float* __restrict__ alpha) {
    __shared__ float ap[Sn];
    __shared__ float gl[66];
    __shared__ unsigned char skipf[Sn];
    const int b = blockIdx.x;
    const int s = threadIdx.x;

    if (s < Sn) {
        bool sk = false;
        if ((s & 1) && s >= 3) { int u = s >> 1; sk = (ys[b * Un + u] != ys[b * Un + u - 1]); }
        skipf[s] = (unsigned char)sk;
    }
    if (s < 65) gl[s] = glogit[(size_t)(b * Tn) * 65 + s];
    if (s == 65) gl[65] = lse[b * Tn];
    __syncthreads();
    if (s < Sn) {
        float v = (s < 2) ? lp_of(gl, s) : -FINF;
        ap[s] = v;
        alpha[((size_t)b * Tn) * Sn + s] = v;
    }
    for (int t = 1; t < Tn; ++t) {
        __syncthreads();                               // (A) ap writes visible, old gl reads done
        float x0 = -FINF, x1 = -FINF, x2 = -FINF;
        if (s < Sn) {
            x0 = ap[s];
            if (s >= 1) x1 = ap[s - 1];
            if (s >= 2 && skipf[s]) x2 = ap[s - 2];
        }
        if (s < 65) gl[s] = glogit[((size_t)b * Tn + t) * 65 + s];
        if (s == 65) gl[65] = lse[b * Tn + t];
        __syncthreads();                               // (B) gl ready, ap reads done
        if (s < Sn) {
            float a = lp_of(gl, s) + safe_la(safe_la(x0, x1), x2);
            ap[s] = a;
            alpha[((size_t)b * Tn + t) * Sn + s] = a;
        }
    }
}

// ---------- kernel 4: beta backward scan (one block per batch) ----------
__global__ __launch_bounds__(256) void beta_kernel(const float* __restrict__ glogit,
                                                   const float* __restrict__ lse,
                                                   const int* __restrict__ ys,
                                                   float* __restrict__ beta) {
    __shared__ float bn_s[Sn];
    __shared__ float mb[Sn];
    __shared__ float gl[66];
    __shared__ unsigned char skipf[Sn];
    const int b = blockIdx.x;
    const int s = threadIdx.x;

    if (s < Sn) {
        bool sk = false;
        if ((s & 1) && s >= 3) { int u = s >> 1; sk = (ys[b * Un + u] != ys[b * Un + u - 1]); }
        skipf[s] = (unsigned char)sk;
    }
    if (s < Sn) {
        float v = (s >= Sn - 2) ? 0.0f : -FINF;
        bn_s[s] = v;
        beta[((size_t)b * Tn + (Tn - 1)) * Sn + s] = v;
    }
    for (int t = Tn - 2; t >= 0; --t) {
        __syncthreads();                               // (A) bn writes visible, old mb/gl reads done
        if (s < 65) gl[s] = glogit[((size_t)b * Tn + t + 1) * 65 + s];
        if (s == 65) gl[65] = lse[b * Tn + t + 1];
        __syncthreads();                               // (B) gl ready
        if (s < Sn) mb[s] = lp_of(gl, s) + bn_s[s];
        __syncthreads();                               // (C) mb ready
        if (s < Sn) {
            float m0 = mb[s];
            float m1 = (s + 1 < Sn) ? mb[s + 1] : -FINF;
            float m2 = (s + 2 < Sn && skipf[s + 2]) ? mb[s + 2] : -FINF;
            float bc = safe_la(safe_la(m0, m1), m2);
            bn_s[s] = bc;
            beta[((size_t)b * Tn + t) * Sn + s] = bc;
        }
    }
}

// ---------- kernel 5: per-(b,u) loss: logsumexp over t of alpha + beta' ----------
__global__ __launch_bounds__(256) void loss_u_kernel(const float* __restrict__ alpha,
                                                     const float* __restrict__ beta,
                                                     const float* __restrict__ glogit,
                                                     const float* __restrict__ lse,
                                                     float* __restrict__ loss_u) {
    const int b = blockIdx.x >> 6;       // 1024 blocks: b = blk/64, u = blk%64
    const int u = blockIdx.x & 63;
    const int s = 2 * u + 1;
    float M = -FINF, Ss = 0.0f;
    for (int t = threadIdx.x; t < Tn; t += 256) {
        const size_t base = ((size_t)b * Tn + t) * Sn + s;
        float av = alpha[base];
        float bpv;
        if (t == Tn - 1) {
            bpv = beta[base];
        } else {
            float bv = beta[base];
            float bnv = beta[base + Sn];
            float pn = glogit[((size_t)b * Tn + t + 1) * 65 + (u + 1)] - lse[b * Tn + t + 1];
            bpv = log_sub_exp(bv, bnv + pn);
        }
        float lsv = av + bpv;
        if (lsv != lsv) lsv = -FINF;     // nan -> -inf (as reference)
        lse_merge(M, Ss, lsv, 1.0f);
    }
    __shared__ float sM[256], sS[256];
    sM[threadIdx.x] = M; sS[threadIdx.x] = Ss;
    __syncthreads();
    for (int off = 128; off > 0; off >>= 1) {
        if (threadIdx.x < off)
            lse_merge(sM[threadIdx.x], sS[threadIdx.x], sM[threadIdx.x + off], sS[threadIdx.x + off]);
        __syncthreads();
    }
    if (threadIdx.x == 0) {
        float r = (sM[0] == -FINF) ? -FINF : sM[0] + logf(sS[0]);
        loss_u[b * Un + u] = r;
    }
}

// ---------- kernel 6: final masked mean -> scalar ----------
__global__ __launch_bounds__(1024) void final_kernel(const float* __restrict__ loss_u,
                                                     const int* __restrict__ hlens,
                                                     float* __restrict__ out) {
    __shared__ float sv[1024];
    __shared__ float sc[1024];
    __shared__ float pb[Bn];
    const int tid = threadIdx.x;
    const float lu = loss_u[tid];
    const bool bad = is_inf_f(lu);
    sv[tid] = bad ? 0.0f : lu;
    sc[tid] = bad ? 0.0f : 1.0f;
    __syncthreads();
    for (int off = 32; off > 0; off >>= 1) {
        if ((tid & 63) < off) { sv[tid] += sv[tid + off]; sc[tid] += sc[tid + off]; }
        __syncthreads();
    }
    if ((tid & 63) == 0) {
        const int b = tid >> 6;
        float lf = sv[tid] / sc[tid];
        if (hlens[b] < Un) lf = 0.0f;
        pb[b] = -lf;
    }
    __syncthreads();
    if (tid == 0) {
        float sacc = 0.0f;
        for (int b2 = 0; b2 < Bn; ++b2) sacc += pb[b2];
        out[0] = sacc / (float)Bn;
    }
}

// ---------- host launcher ----------
extern "C" void kernel_launch(void* const* d_in, const int* in_sizes, int n_in,
                              void* d_out, int out_size, void* d_ws, size_t ws_size,
                              hipStream_t stream) {
    (void)in_sizes; (void)n_in; (void)out_size; (void)ws_size;
    const float* hs    = (const float*)d_in[0];  // [B,T,D]
    const int*   hlens = (const int*)d_in[1];    // [B]
    const int*   ys    = (const int*)d_in[2];    // [B,U]
    /* ali = d_in[3] unused (risk_factor == 0) */
    const float* W     = (const float*)d_in[4];  // [D,V]
    const float* bias  = (const float*)d_in[5];  // [V]
    float* out = (float*)d_out;

    char* ws = (char*)d_ws;
    _Float16* Wt     = (_Float16*)(ws + 0);                 //  5,128,192 B: [VPAD][D] f16
    float*    lse_p  = (float*)  (ws + 5128192);            //     64,000 B: [B*T]
    float*    glog_p = (float*)  (ws + 5192192);            //  4,160,000 B: [B*T][65]
    float*    alph_p = (float*)  (ws + 9352192);            //  8,256,000 B: [B][T][S]
    float*    beta_p = (float*)  (ws + 17608192);           //  8,256,000 B: [B][T][S]
    float*    lu_p   = (float*)  (ws + 25864192);           //      4,096 B: [B*U]
    float*    biasp  = (float*)  (ws + 25868288);           //     20,032 B: [VPAD]

    prep_wt_kernel<<<(VPAD * Dn + 255) / 256, 256, 0, stream>>>(W, bias, Wt, biasp);
    gemm_lse_kernel<<<Bn * Tn / 16, 256, 0, stream>>>(hs, Wt, biasp, lse_p);
    gather_logits_kernel<<<Bn * Tn, 128, 0, stream>>>(hs, Wt, bias, ys, glog_p);
    alpha_kernel<<<Bn, 256, 0, stream>>>(glog_p, lse_p, ys, alph_p);
    beta_kernel<<<Bn, 256, 0, stream>>>(glog_p, lse_p, ys, beta_p);
    loss_u_kernel<<<Bn * Un, 256, 0, stream>>>(alph_p, beta_p, glog_p, lse_p, lu_p);
    final_kernel<<<1, 1024, 0, stream>>>(lu_p, hlens, out);
}
